// LSTMDecoder_50173807952074
// MI455X (gfx1250) — compile-verified, matrix operand order checked by reading, not run on hardware
//
#include <hip/hip_runtime.h>

#define LATENT 64
#define HID    512
#define OBS    32
#define SEQ    70
#define TILE_B 64
#define NTHREADS 512   // 16 waves of 32 (wave32)

typedef __attribute__((ext_vector_type(8)))  __bf16 v8bf;
typedef __attribute__((ext_vector_type(16))) __bf16 v16bf;
typedef __attribute__((ext_vector_type(8)))  float  v8f;

// ---------------- helpers ----------------

__device__ __forceinline__ unsigned short f2bf(float f) {
    union { float f; unsigned int u; } v; v.f = f;
    unsigned int u = v.u;
    unsigned int r = u + 0x7FFFu + ((u >> 16) & 1u);   // round-to-nearest-even
    return (unsigned short)(r >> 16);
}

// Load one WMMA 16-bit fragment for this lane.
// For both A (16x32) and B (32x16) on CDNA5 the per-lane data is two
// contiguous 8-element (16B) chunks of the row/column at k0 and k0+16,
// with k0 = kblock*32 + (lane>>4)*8. Works for LDS or global pointers.
__device__ __forceinline__ v16bf load_frag(const unsigned short* row, int k0) {
    v8bf lo = *reinterpret_cast<const v8bf*>(row + k0);
    v8bf hi = *reinterpret_cast<const v8bf*>(row + k0 + 16);
    return __builtin_shufflevector(lo, hi, 0,1,2,3,4,5,6,7,8,9,10,11,12,13,14,15);
}

__device__ __forceinline__ v8f wmma_bf16(v16bf a, v16bf b, v8f c) {
    // v_wmma_f32_16x16x32_bf16 : D = A*B + C, f32 accumulate
    return __builtin_amdgcn_wmma_f32_16x16x32_bf16(false, a, false, b,
                                                   (short)0, c, false, false);
}

__device__ __forceinline__ float fsigmoid(float x) {
    return __builtin_amdgcn_rcpf(1.0f + __expf(-x));
}
__device__ __forceinline__ float ftanh_fast(float x) {
    // numerically safe tanh via sigmoid
    return 2.0f * fsigmoid(2.0f * x) - 1.0f;
}

// ---------------- pre-pass kernels ----------------

__global__ void f32_to_bf16_kernel(const float* __restrict__ src,
                                   unsigned short* __restrict__ dst, int n) {
    int i = blockIdx.x * blockDim.x + threadIdx.x;
    int stride = gridDim.x * blockDim.x;
    for (; i < n; i += stride) dst[i] = f2bf(src[i]);
}

__global__ void bias_combine_kernel(const float* __restrict__ a,
                                    const float* __restrict__ b,
                                    float* __restrict__ dst, int n) {
    int i = blockIdx.x * blockDim.x + threadIdx.x;
    if (i < n) dst[i] = a[i] + b[i];
}

// ---------------- main persistent LSTM kernel ----------------

__global__ __launch_bounds__(NTHREADS)
void lstm_decoder_kernel(const unsigned short* __restrict__ zb,    // [B,64]  bf16
                         const unsigned short* __restrict__ Wlh,   // [512,64] bf16
                         const float*          __restrict__ blh,   // [512]
                         const unsigned short* __restrict__ Wlc,   // [512,64] bf16
                         const float*          __restrict__ blc,   // [512]
                         const unsigned short* __restrict__ Wih,   // [2048,32] bf16
                         const unsigned short* __restrict__ Whh,   // [2048,512] bf16
                         const float*          __restrict__ bcomb, // [2048] = b_ih+b_hh
                         const unsigned short* __restrict__ Wout,  // [32,512] bf16
                         const float*          __restrict__ bout,  // [32]
                         float*                __restrict__ out)   // [B,SEQ,OBS]
{
    extern __shared__ unsigned char smem_raw[];
    unsigned short* hb0 = (unsigned short*)smem_raw;             // 64*512 bf16  (64KB)
    unsigned short* hb1 = hb0 + TILE_B * HID;                    // 64*512 bf16  (64KB)
    float*          cls = (float*)(hb1 + TILE_B * HID);          // 64*512 f32  (128KB)
    unsigned short* xb  = (unsigned short*)(cls + TILE_B * HID); // 64*32  bf16   (4KB)

    const int tid  = threadIdx.x;
    const int lane = tid & 31;
    const int wave = tid >> 5;
    const int lm   = lane & 15;   // row (A) / col (B,C,D) within tile
    const int sel  = lane >> 4;   // K-chunk / M-half selector
    const int b0   = blockIdx.x * TILE_B;

    // Warm L2/WGP$ with the recurrent weight matrix (dominant operand).
    for (int off = tid * 256; off < 4 * HID * HID * 2; off += NTHREADS * 256)
        __builtin_prefetch((const char*)Whh + off, 0, 1);

    // x0 = 0
    for (int i = tid; i < TILE_B * OBS; i += NTHREADS) xb[i] = 0;

    // ---- init: h0 = z @ Wlh^T + blh ; c0 = z @ Wlc^T + blc ----
    // One wave per u-block pair; B-fragments (Wlh/Wlc) reused across 4 batch blocks.
    for (int uu = wave; uu < HID / 16; uu += NTHREADS / 32) {
        const int u0 = uu * 16;
        const float bh = blh[u0 + lm];
        const float bc = blc[u0 + lm];
        v8f ah[4], ac[4];
        #pragma unroll
        for (int bb = 0; bb < 4; ++bb) {
            v8f th = {bh, bh, bh, bh, bh, bh, bh, bh};
            v8f tc = {bc, bc, bc, bc, bc, bc, bc, bc};
            ah[bb] = th; ac[bb] = tc;
        }
        const unsigned short* whr = Wlh + (u0 + lm) * LATENT;
        const unsigned short* wcr = Wlc + (u0 + lm) * LATENT;
        #pragma unroll
        for (int kb = 0; kb < LATENT / 32; ++kb) {
            const int k0 = kb * 32 + sel * 8;
            v16bf bwh = load_frag(whr, k0);
            v16bf bwc = load_frag(wcr, k0);
            #pragma unroll
            for (int bb = 0; bb < 4; ++bb) {
                v16bf a = load_frag(zb + (size_t)(b0 + bb * 16 + lm) * LATENT, k0);
                ah[bb] = wmma_bf16(a, bwh, ah[bb]);
                ac[bb] = wmma_bf16(a, bwc, ac[bb]);
            }
        }
        #pragma unroll
        for (int bb = 0; bb < 4; ++bb) {
            #pragma unroll
            for (int r = 0; r < 8; ++r) {
                const int m = bb * 16 + sel * 8 + r;
                hb0[m * HID + u0 + lm] = f2bf(ah[bb][r]);
                cls[m * HID + u0 + lm] = ac[bb][r];
            }
        }
    }
    __syncthreads();

    // ---- recurrence ----
    for (int t = 0; t < SEQ; ++t) {
        unsigned short* hcur = (t & 1) ? hb1 : hb0;
        unsigned short* hnxt = (t & 1) ? hb0 : hb1;

        // phase 1: gates = x@Wih^T + h@Whh^T + (b_ih+b_hh); fused cell update.
        // Each wave owns a u-block; W fragments are loaded once per k-step and
        // reused across all 4 batch blocks (16 back-to-back WMMAs per k-step).
        for (int uu = wave; uu < HID / 16; uu += NTHREADS / 32) {
            const int u0 = uu * 16;

            v8f acc[4][4];   // [batch-block][gate]
            #pragma unroll
            for (int g = 0; g < 4; ++g) {
                const float bv = bcomb[g * HID + u0 + lm];
                #pragma unroll
                for (int bb = 0; bb < 4; ++bb) {
                    v8f tmp = {bv, bv, bv, bv, bv, bv, bv, bv};
                    acc[bb][g] = tmp;
                }
            }

            // x contribution (K = OBS = 32, one WMMA per (bb, gate))
            {
                v16bf bx[4];
                #pragma unroll
                for (int g = 0; g < 4; ++g)
                    bx[g] = load_frag(Wih + (size_t)(g * HID + u0 + lm) * OBS, sel * 8);
                #pragma unroll
                for (int bb = 0; bb < 4; ++bb) {
                    v16bf ax = load_frag(xb + (bb * 16 + lm) * OBS, sel * 8);
                    #pragma unroll
                    for (int g = 0; g < 4; ++g)
                        acc[bb][g] = wmma_bf16(ax, bx[g], acc[bb][g]);
                }
            }

            // h contribution (K = HID = 512 -> 16 k-steps)
            #pragma unroll 2
            for (int kb = 0; kb < HID / 32; ++kb) {
                const int k0 = kb * 32 + sel * 8;
                v16bf bw[4];
                #pragma unroll
                for (int g = 0; g < 4; ++g)
                    bw[g] = load_frag(Whh + (size_t)(g * HID + u0 + lm) * HID, k0);
                #pragma unroll
                for (int bb = 0; bb < 4; ++bb) {
                    v16bf a = load_frag(hcur + (bb * 16 + lm) * HID, k0);
                    #pragma unroll
                    for (int g = 0; g < 4; ++g)
                        acc[bb][g] = wmma_bf16(a, bw[g], acc[bb][g]);
                }
            }

            // LSTM cell update (i,f,g,o quarters, PyTorch order)
            #pragma unroll
            for (int bb = 0; bb < 4; ++bb) {
                #pragma unroll
                for (int r = 0; r < 8; ++r) {
                    const int m = bb * 16 + sel * 8 + r;
                    float i_ = fsigmoid(acc[bb][0][r]);
                    float f_ = fsigmoid(acc[bb][1][r]);
                    float g_ = ftanh_fast(acc[bb][2][r]);
                    float o_ = fsigmoid(acc[bb][3][r]);
                    float cold = cls[m * HID + u0 + lm];
                    float cnew = f_ * cold + i_ * g_;
                    cls[m * HID + u0 + lm] = cnew;
                    hnxt[m * HID + u0 + lm] = f2bf(o_ * ftanh_fast(cnew));
                }
            }
        }
        __syncthreads();

        // phase 2: out_t = h_new @ Wout^T + bout ; feed back x (bf16)
        if (wave < (TILE_B / 16) * (OBS / 16)) {   // 8 tiles, waves 0..7
            const int bb = wave >> 1;
            const int n0 = (wave & 1) * 16;
            const float bv = bout[n0 + lm];
            v8f acc = {bv, bv, bv, bv, bv, bv, bv, bv};
            const unsigned short* hrow = hnxt + (bb * 16 + lm) * HID;
            const unsigned short* wrow = Wout + (n0 + lm) * HID;
            #pragma unroll 4
            for (int kb = 0; kb < HID / 32; ++kb) {
                const int k0 = kb * 32 + sel * 8;
                acc = wmma_bf16(load_frag(hrow, k0), load_frag(wrow, k0), acc);
            }
            #pragma unroll
            for (int r = 0; r < 8; ++r) {
                const int m = bb * 16 + sel * 8 + r;
                float v = acc[r];
                out[((size_t)(b0 + m) * SEQ + t) * OBS + n0 + lm] = v;
                xb[m * OBS + n0 + lm] = f2bf(v);
            }
        }
        __syncthreads();
    }
}

// ---------------- host launcher ----------------

extern "C" void kernel_launch(void* const* d_in, const int* in_sizes, int n_in,
                              void* d_out, int out_size, void* d_ws, size_t ws_size,
                              hipStream_t stream) {
    const float* z     = (const float*)d_in[0];
    const float* W_lh  = (const float*)d_in[1];
    const float* b_lh  = (const float*)d_in[2];
    const float* W_lc  = (const float*)d_in[3];
    const float* b_lc  = (const float*)d_in[4];
    const float* W_ih  = (const float*)d_in[5];
    const float* W_hh  = (const float*)d_in[6];
    const float* b_ih  = (const float*)d_in[7];
    const float* b_hh  = (const float*)d_in[8];
    const float* W_out = (const float*)d_in[9];
    const float* b_out = (const float*)d_in[10];

    const int B = in_sizes[0] / LATENT;   // 32768

    // workspace layout (bf16 copies + combined bias), all 16B+ aligned
    unsigned short* zb    = (unsigned short*)d_ws;
    unsigned short* Wlhb  = zb    + (size_t)B * LATENT;       // 512*64
    unsigned short* Wlcb  = Wlhb  + HID * LATENT;             // 512*64
    unsigned short* Wihb  = Wlcb  + HID * LATENT;             // 2048*32
    unsigned short* Whhb  = Wihb  + 4 * HID * OBS;            // 2048*512
    unsigned short* Woutb = Whhb  + 4 * HID * HID;            // 32*512
    float*          bcomb = (float*)(Woutb + OBS * HID);      // 2048

    // f32 -> bf16 conversions
    {
        int n;
        n = B * LATENT;
        f32_to_bf16_kernel<<<(n + 255) / 256, 256, 0, stream>>>(z, zb, n);
        n = HID * LATENT;
        f32_to_bf16_kernel<<<(n + 255) / 256, 256, 0, stream>>>(W_lh, Wlhb, n);
        f32_to_bf16_kernel<<<(n + 255) / 256, 256, 0, stream>>>(W_lc, Wlcb, n);
        n = 4 * HID * OBS;
        f32_to_bf16_kernel<<<(n + 255) / 256, 256, 0, stream>>>(W_ih, Wihb, n);
        n = 4 * HID * HID;
        f32_to_bf16_kernel<<<(n + 255) / 256, 256, 0, stream>>>(W_hh, Whhb, n);
        n = OBS * HID;
        f32_to_bf16_kernel<<<(n + 255) / 256, 256, 0, stream>>>(W_out, Woutb, n);
        n = 4 * HID;
        bias_combine_kernel<<<(n + 255) / 256, 256, 0, stream>>>(b_ih, b_hh, bcomb, n);
    }

    // persistent LSTM: one workgroup per 64 batch rows; state lives in LDS.
    // LDS: 2*64*512*2 (h dbl-buf) + 64*512*4 (c) + 64*32*2 (x) = 266240 B < 320KB/WGP
    size_t lds_bytes = (size_t)(2 * TILE_B * HID * 2 + TILE_B * HID * 4 + TILE_B * OBS * 2);
    lstm_decoder_kernel<<<B / TILE_B, NTHREADS, lds_bytes, stream>>>(
        zb, Wlhb, b_lh, Wlcb, b_lc, Wihb, Whhb, bcomb, Woutb, b_out,
        (float*)d_out);
}
